// My_GRU_Layer_43928925504256
// MI455X (gfx1250) — matrix operand order repaired and checked by reading, not compiled, not run on hardware
//
#include <hip/hip_runtime.h>

typedef __attribute__((ext_vector_type(16))) __bf16 v16bf;
typedef __attribute__((ext_vector_type(8)))  float  v8f;

#define VOCAB 5000
#define HID   256
#define H3    768
#define BATCH 64
#define STEPS 256
#define NT_PAD 320                         // ceil(5000/16)=313 padded to 320 tiles
#define OUT_ELEMS ((size_t)STEPS * BATCH * VOCAB)   // 81,920,000

// ---------------------------------------------------------------------------
// Pre-swizzle rkernel [256,768] f32 -> bf16 B-fragments [48 ntile][8 kb][32 lane][16]
// B frag (32x16 bf16): lane<16: elem i = B[kb*32+i+(i>=8?8:0)][nt*16+lane]
//                      lane>=16: elem i = B[kb*32+8+i+(i>=8?8:0)][nt*16+lane-16]
// ---------------------------------------------------------------------------
__global__ void prep_rk_frag(const float* __restrict__ rk, __bf16* __restrict__ rkf) {
    int idx = blockIdx.x * 256 + threadIdx.x;          // 48*8*32 = 12288
    if (idx >= 48 * 8 * 32) return;
    int lane = idx & 31;
    int kb   = (idx >> 5) & 7;
    int nt   = idx >> 8;
    int n    = nt * 16 + (lane & 15);
    int kbase = kb * 32 + ((lane < 16) ? 0 : 8);
    union { v16bf v; __bf16 e[16]; } uf;
#pragma unroll
    for (int i = 0; i < 16; ++i)
        uf.e[i] = (__bf16)rk[(size_t)(kbase + i + ((i >= 8) ? 8 : 0)) * H3 + n];
    *(v16bf*)(rkf + (size_t)idx * 16) = uf.v;
}

// Wd [256,5000] f32 -> bf16 B-fragments [320][8][32][16], zero-padded cols >= 5000
__global__ void prep_wd_frag(const float* __restrict__ wd, __bf16* __restrict__ wdf) {
    int idx = blockIdx.x * 256 + threadIdx.x;          // 320*8*32 = 81920
    if (idx >= NT_PAD * 8 * 32) return;
    int lane = idx & 31;
    int kb   = (idx >> 5) & 7;
    int nt   = idx >> 8;
    int n    = nt * 16 + (lane & 15);
    int kbase = kb * 32 + ((lane < 16) ? 0 : 8);
    union { v16bf v; __bf16 e[16]; } uf;
#pragma unroll
    for (int i = 0; i < 16; ++i)
        uf.e[i] = (n < VOCAB) ? (__bf16)wd[(size_t)(kbase + i + ((i >= 8) ? 8 : 0)) * VOCAB + n] : (__bf16)0.0f;
    *(v16bf*)(wdf + (size_t)idx * 16) = uf.v;
}

// ---------------------------------------------------------------------------
// GRU scan. 4 blocks x 512 threads; block handles 16 batch rows independently.
// rkernel B-fragments stay resident in VGPRs across all 256 steps (compiler
// hoists the loop-invariant global loads); per step: LDS A-frag loads + 24 WMMA.
// ---------------------------------------------------------------------------
__global__ __launch_bounds__(512) void gru_scan(
    const int* __restrict__ tokens, const float* __restrict__ h0,
    const float* __restrict__ kern, const float* __restrict__ bias,
    const __bf16* __restrict__ rkf, __bf16* __restrict__ yfrag,
    float* __restrict__ hlast)
{
    __shared__ float hbuf[16][260];                    // fp32 hidden state (padded)
    __shared__ float mh[16][772];                      // matmul result (padded)
    __shared__ __align__(32) __bf16 hfrag[8][32][16];  // A-fragments of h

    const int tid  = threadIdx.x;
    const int b0   = blockIdx.x * 16;
    const int lane = tid & 31;
    const int wave = tid >> 5;

    for (int e = tid; e < 16 * HID; e += 512) {
        int r = e >> 8, j = e & 255;
        hbuf[r][j] = h0[(b0 + r) * HID + j];
    }
    __syncthreads();
    if (tid < 256) {                                   // initial A-fragments from h0
        int kb = tid >> 5, l = tid & 31;
        int row  = l & 15;
        int koff = kb * 32 + ((l < 16) ? 0 : 8);
        union { v16bf v; __bf16 e[16]; } uf;
#pragma unroll
        for (int i = 0; i < 8; ++i) {
            uf.e[i]     = (__bf16)hbuf[row][koff + i];
            uf.e[i + 8] = (__bf16)hbuf[row][koff + 16 + i];
        }
        *(v16bf*)&hfrag[kb][l][0] = uf.v;
    }
    __syncthreads();

    for (int t = 0; t < STEPS; ++t) {
        // -------- phase 1: mh = h @ rkernel (bf16 WMMA, f32 accum) --------
        v8f acc0 = {}, acc1 = {}, acc2 = {};
        const int nt0 = wave * 3;                      // 16 waves x 3 tiles = 48
#pragma unroll
        for (int kb = 0; kb < 8; ++kb) {
            v16bf a = *(const v16bf*)&hfrag[kb][lane][0];
            const __bf16* bp = rkf + (((size_t)(nt0 * 8 + kb)) * 32 + lane) * 16;
            v16bf bA = *(const v16bf*)(bp);
            v16bf bB = *(const v16bf*)(bp + 4096);     // next ntile: 8*32*16 elems
            v16bf bC = *(const v16bf*)(bp + 8192);
            acc0 = __builtin_amdgcn_wmma_f32_16x16x32_bf16(false, a, false, bA, (short)0, acc0, false, false);
            acc1 = __builtin_amdgcn_wmma_f32_16x16x32_bf16(false, a, false, bB, (short)0, acc1, false, false);
            acc2 = __builtin_amdgcn_wmma_f32_16x16x32_bf16(false, a, false, bC, (short)0, acc2, false, false);
        }
        {
            const int rowoff = (lane >> 4) * 8;        // C layout: lanes16-31 -> M+8
            const int cl = lane & 15;
            union { v8f v; float f[8]; } u0, u1, u2;
            u0.v = acc0; u1.v = acc1; u2.v = acc2;
#pragma unroll
            for (int r = 0; r < 8; ++r) {
                mh[r + rowoff][(nt0 + 0) * 16 + cl] = u0.f[r];
                mh[r + rowoff][(nt0 + 1) * 16 + cl] = u1.f[r];
                mh[r + rowoff][(nt0 + 2) * 16 + cl] = u2.f[r];
            }
        }
        __syncthreads();

        // -------- phase 2: gates (fp32, exact one-hot gather) --------
        for (int e = tid; e < 16 * HID; e += 512) {
            int r = e >> 8, j = e & 255;
            int tok = tokens[(b0 + r) * STEPS + t];
            const float* krow = kern + (size_t)tok * H3;
            float xz = krow[j]       + bias[j];
            float xr = krow[256 + j] + bias[256 + j];
            float xh = krow[512 + j] + bias[512 + j];
            float hz = mh[r][j]       + bias[768  + j];
            float hr = mh[r][256 + j] + bias[1024 + j];
            float hh = mh[r][512 + j] + bias[1280 + j];
            float z    = 1.0f / (1.0f + __expf(-(xz + hz)));
            float rr   = 1.0f / (1.0f + __expf(-(xr + hr)));
            float cand = tanhf(xh + rr * hh);
            float hold = hbuf[r][j];
            hbuf[r][j] = z * hold + (1.0f - z) * cand;
        }
        __syncthreads();

        // -------- phase 3: next-step A-fragments + Y fragment tile --------
        if (tid < 256) {
            int kb = tid >> 5, l = tid & 31;
            int row  = l & 15;
            int koff = kb * 32 + ((l < 16) ? 0 : 8);
            union { v16bf v; __bf16 e[16]; } uf;
#pragma unroll
            for (int i = 0; i < 8; ++i) {
                uf.e[i]     = (__bf16)hbuf[row][koff + i];
                uf.e[i + 8] = (__bf16)hbuf[row][koff + 16 + i];
            }
            *(v16bf*)&hfrag[kb][l][0] = uf.v;
            const int mt = t * 4 + blockIdx.x;         // M-tile = rows t*64+b0..+15
            *(v16bf*)(yfrag + (((size_t)(mt * 8 + kb)) * 32 + l) * 16) = uf.v;
        }
        __syncthreads();
    }

    for (int e = tid; e < 16 * HID; e += 512) {        // h_last -> d_out tail
        int r = e >> 8, j = e & 255;
        hlast[(b0 + r) * HID + j] = hbuf[r][j];
    }
}

// ---------------------------------------------------------------------------
// Output GEMM: out[16384,5000] = Y @ Wd + bd.
// 256 threads = 8 waves as 2 (M) x 4 (N); each wave computes 2 M-tiles x
// 4 N-tiles (32x64) -> 6 fragment loads per 8 WMMAs. Block = 64 rows x 256 cols.
// Output written with non-temporal stores (write-once stream, keep L2 for frags).
// ---------------------------------------------------------------------------
__global__ __launch_bounds__(256) void out_gemm(
    const __bf16* __restrict__ yfrag, const __bf16* __restrict__ wdf,
    const float* __restrict__ bd, float* __restrict__ out)
{
    const int tid  = threadIdx.x;
    const int lane = tid & 31;
    const int wave = tid >> 5;
    const int mt0  = blockIdx.y * 4 + (wave >> 2) * 2; // 0..1022 (2 tiles per wave)
    const int nt0  = blockIdx.x * 16 + (wave & 3) * 4; // 0..316

    v8f accA[4] = {};
    v8f accB[4] = {};
#pragma unroll
    for (int kb = 0; kb < 8; ++kb) {
        v16bf a0 = *(const v16bf*)(yfrag + (((size_t)((mt0 + 0) * 8 + kb)) * 32 + lane) * 16);
        v16bf a1 = *(const v16bf*)(yfrag + (((size_t)((mt0 + 1) * 8 + kb)) * 32 + lane) * 16);
#pragma unroll
        for (int n = 0; n < 4; ++n) {
            v16bf b = *(const v16bf*)(wdf + (((size_t)((nt0 + n) * 8 + kb)) * 32 + lane) * 16);
            accA[n] = __builtin_amdgcn_wmma_f32_16x16x32_bf16(false, a0, false, b, (short)0, accA[n], false, false);
            accB[n] = __builtin_amdgcn_wmma_f32_16x16x32_bf16(false, a1, false, b, (short)0, accB[n], false, false);
        }
    }

    const int cl = lane & 15;
    const int rh = (lane < 16) ? 0 : 8;                // C layout: lanes16-31 -> M+8
#pragma unroll
    for (int n = 0; n < 4; ++n) {
        int col = (nt0 + n) * 16 + cl;
        if (col < VOCAB) {
            float bdv = bd[col];
            union { v8f v; float f[8]; } uA, uB;
            uA.v = accA[n]; uB.v = accB[n];
            const int rbaseA = (mt0 + 0) * 16 + rh;
            const int rbaseB = (mt0 + 1) * 16 + rh;
#pragma unroll
            for (int r = 0; r < 8; ++r) {
                __builtin_nontemporal_store(uA.f[r] + bdv, out + (size_t)(rbaseA + r) * VOCAB + col);
                __builtin_nontemporal_store(uB.f[r] + bdv, out + (size_t)(rbaseB + r) * VOCAB + col);
            }
        }
    }
}

// ---------------------------------------------------------------------------
extern "C" void kernel_launch(void* const* d_in, const int* in_sizes, int n_in,
                              void* d_out, int out_size, void* d_ws, size_t ws_size,
                              hipStream_t stream) {
    const int*   tokens = (const int*)  d_in[0];   // [64,256]
    const float* h0     = (const float*)d_in[1];   // [64,256]
    const float* kern   = (const float*)d_in[2];   // [5000,768]
    const float* rk     = (const float*)d_in[3];   // [256,768]
    const float* bias   = (const float*)d_in[4];   // [2,768]
    const float* wd     = (const float*)d_in[5];   // [256,5000]
    const float* bd     = (const float*)d_in[6];   // [5000]
    float* out = (float*)d_out;

    char* ws = (char*)d_ws;
    __bf16* rkf = (__bf16*)(ws);                              //   393,216 B
    __bf16* wdf = (__bf16*)(ws + 393216);                     // 2,621,440 B
    __bf16* yfr = (__bf16*)(ws + 393216 + 2621440);           // 8,388,608 B

    prep_rk_frag<<<48, 256, 0, stream>>>(rk, rkf);
    prep_wd_frag<<<NT_PAD * 8 * 32 / 256, 256, 0, stream>>>(wd, wdf);
    gru_scan<<<4, 512, 0, stream>>>(tokens, h0, kern, bias, rkf, yfr, out + OUT_ELEMS);
    out_gemm<<<dim3(20, 256), 256, 0, stream>>>(yfr, wdf, bd, out);
}
